// MACCell_17583596109909
// MI455X (gfx1250) — compile-verified, matrix operand order checked by reading, not run on hardware
//
#include <hip/hip_runtime.h>
#include <hip/hip_bf16.h>

typedef __bf16 bf16;
typedef __attribute__((ext_vector_type(16))) __bf16 v16bf;
typedef __attribute__((ext_vector_type(8)))  __bf16 v8bf;
typedef __attribute__((ext_vector_type(8)))  float  v8f;

#define WMMA_BF16(a,b,c) __builtin_amdgcn_wmma_f32_16x16x32_bf16(false,(a),false,(b),(short)0,(c),false,false)

__device__ __forceinline__ v8f vzero8(){
  v8f z;
  #pragma unroll
  for (int i=0;i<8;++i) z[i]=0.f;
  return z;
}

// WMMA 16-bit operand layout per lane: elements 0..7 = K sel+0..7, elements 8..15 = K sel+16..23
// (sel = 0 for lanes 0-15, 8 for lanes 16-31; caller folds sel into p).
__device__ __forceinline__ v16bf ld_op_bf16(const bf16* p){
  const v8bf* q = (const v8bf*)p;
  v8bf lo = q[0], hi = q[2];
  v16bf r;
  #pragma unroll
  for (int i=0;i<8;++i){ r[i]=lo[i]; r[i+8]=hi[i]; }
  return r;
}

__device__ __forceinline__ v16bf ld_op_f32(const float* p){
  v16bf r;
  #pragma unroll
  for (int i=0;i<8;++i){ r[i]=(bf16)p[i]; r[i+8]=(bf16)p[i+16]; }
  return r;
}

// dst[n*K + k] = bf16(src[k*N + n] (+ src2[k*N + n]))   (src: K x N f32 row-major)
__global__ __launch_bounds__(256) void transpose_cvt_kernel(
    const float* __restrict__ src, const float* __restrict__ src2,
    bf16* __restrict__ dst, int K, int N){
  int tot = K*N;
  for (int idx = blockIdx.x*256 + threadIdx.x; idx < tot; idx += gridDim.x*256){
    int n = idx / K, k = idx - n*K;
    float v = src[(size_t)k*N + n];
    if (src2) v += src2[(size_t)k*N + n];
    dst[idx] = (bf16)v;
  }
}

// C = act(A0@W0 + A1@W1 + bias), A f32 (MxK row-major), W bf16 stored transposed (N x K row-major).
// One wave per 16x16 C tile. act: 0=none, 1=tanh.
__global__ __launch_bounds__(256) void gemm2_wmma_kernel(
    const float* __restrict__ A0, const bf16* __restrict__ W0, int K0,
    const float* __restrict__ A1, const bf16* __restrict__ W1, int K1,
    const float* __restrict__ bias, float* __restrict__ C, int M, int N, int act){
  const int lane = threadIdx.x & 31, wave = threadIdx.x >> 5;
  const int lidx = lane & 15, half8 = (lane >> 4) * 8;
  const int ntiles_n = N >> 4;
  const int tiles = (M >> 4) * ntiles_n;
  const int nwaves = gridDim.x * 8;
  for (int t = blockIdx.x*8 + wave; t < tiles; t += nwaves){
    int mt = t / ntiles_n, nt = t - mt*ntiles_n;
    int arow = mt*16 + lidx;
    int wrow = nt*16 + lidx;
    v8f acc = vzero8();
    for (int kb = 0; kb < K0; kb += 32){
      v16bf a = ld_op_f32 (A0 + (size_t)arow*K0 + kb + half8);
      v16bf b = ld_op_bf16(W0 + (size_t)wrow*K0 + kb + half8);
      acc = WMMA_BF16(a,b,acc);
    }
    if (A1){
      for (int kb = 0; kb < K1; kb += 32){
        v16bf a = ld_op_f32 (A1 + (size_t)arow*K1 + kb + half8);
        v16bf b = ld_op_bf16(W1 + (size_t)wrow*K1 + kb + half8);
        acc = WMMA_BF16(a,b,acc);
      }
    }
    int n = nt*16 + lidx;
    float bv = bias ? bias[n] : 0.f;
    #pragma unroll
    for (int r=0;r<8;++r){
      int m = mt*16 + half8 + r;
      float v = acc[r] + bv;
      if (act==1) v = tanhf(v);
      C[(size_t)m*N + n] = v;
    }
  }
}

// Control attention over S=32 words: logit[s] = sum_d cq[b,d]*wac[d]*words[b,s,d] (bac cancels)
__global__ __launch_bounds__(256) void ctrl_attn_kernel(
    const float* __restrict__ words, const float* __restrict__ cq,
    const float* __restrict__ wac, float* __restrict__ control){
  const int b = blockIdx.x, tid = threadIdx.x;
  const int lane = tid & 31, wave = tid >> 5;
  __shared__ float u[512];
  __shared__ float lg[32];
  __shared__ float aw[32];
  for (int i=tid;i<512;i+=256) u[i] = cq[(size_t)b*512+i]*wac[i];
  __syncthreads();
  #pragma unroll
  for (int si=0; si<4; ++si){
    int s = wave*4 + si;
    const float* wp = words + ((size_t)b*32 + s)*512;
    float p = 0.f;
    for (int d=lane; d<512; d+=32) p += u[d]*wp[d];
    #pragma unroll
    for (int m=1;m<32;m<<=1) p += __shfl_xor(p, m, 32);
    if (lane==0) lg[s]=p;
  }
  __syncthreads();
  if (tid==0){
    float mx=-1e30f;
    for(int s=0;s<32;++s) mx=fmaxf(mx,lg[s]);
    float sm=0.f;
    for(int s=0;s<32;++s){ float e=__expf(lg[s]-mx); aw[s]=e; sm+=e; }
    float inv=1.f/sm;
    for(int s=0;s<32;++s) aw[s]*=inv;
  }
  __syncthreads();
  for (int d=tid; d<512; d+=256){
    float a=0.f;
    #pragma unroll 4
    for(int s=0;s<32;++s) a += aw[s]*words[((size_t)b*32+s)*512 + d];
    control[(size_t)b*512+d] = a;
  }
}

// pk[e,n] = sum_d Wk[d,e]*know[b,d,n] + bk[e];  writes pkT[b][n][e] and pk1T = pm[e]*pkT
// 4 blocks per batch (e-range 128 each). know staged transposed+bf16 in LDS in 64-deep K slabs.
__global__ __launch_bounds__(256) void pk_wmma_kernel(
    const float* __restrict__ know, const bf16* __restrict__ WkT,
    const float* __restrict__ pm, const float* __restrict__ bk,
    bf16* __restrict__ pkT, bf16* __restrict__ pk1T){
  const int b = blockIdx.x >> 2, eblk = blockIdx.x & 3;
  const int tid = threadIdx.x, lane = tid & 31, wave = tid >> 5;
  const int lidx = lane & 15, half8 = (lane >> 4)*8;
  __shared__ __align__(16) bf16 klds[208*64];   // [n][d] transposed slab
  __shared__ float pml[512], bkl[512];
  for (int i=tid;i<512;i+=256){ pml[i]=pm[(size_t)b*512+i]; bkl[i]=bk[i]; }
  const int Eb = (eblk*8 + wave)*16;            // this wave's e-tile base
  v8f acc[13];
  #pragma unroll
  for (int t=0;t<13;++t) acc[t]=vzero8();
  for (int slab=0; slab<8; ++slab){
    __syncthreads();
    for (int idx=tid; idx<208*64; idx+=256){
      int n = idx % 208, dd = idx / 208;
      float v = (n<196) ? know[((size_t)b*512 + slab*64 + dd)*196 + n] : 0.f;
      klds[n*64 + dd] = (bf16)v;
    }
    __syncthreads();
    #pragma unroll
    for (int ks=0; ks<2; ++ks){
      v16bf a = ld_op_bf16(WkT + (size_t)(Eb + lidx)*512 + slab*64 + ks*32 + half8);
      #pragma unroll
      for (int nt=0; nt<13; ++nt){
        v16bf bop = ld_op_bf16(klds + (nt*16 + lidx)*64 + ks*32 + half8);
        acc[nt] = WMMA_BF16(a, bop, acc[nt]);
      }
    }
  }
  #pragma unroll
  for (int nt=0; nt<13; ++nt){
    int n = nt*16 + lidx;
    int eb2 = Eb + half8;
    v8bf o1, o2;
    #pragma unroll
    for (int r=0;r<8;++r){
      float v = acc[nt][r] + bkl[eb2+r];
      o1[r] = (bf16)v;
      o2[r] = (bf16)(v * pml[eb2+r]);
    }
    size_t off = ((size_t)b*208 + n)*512 + eb2;
    *(v8bf*)(pkT  + off) = o1;
    *(v8bf*)(pk1T + off) = o2;
  }
}

// Fused: h = relu(pk1T@Wc1a + pkT@Wc1b + c3); h2 = h@Wc2; logit[n]=sum h2*cw; softmax; read.
__global__ __launch_bounds__(256) void mh_wmma_kernel(
    const bf16* __restrict__ pkT, const bf16* __restrict__ pk1T,
    const bf16* __restrict__ Wc1aT, const bf16* __restrict__ Wc1bT,
    const bf16* __restrict__ Wc2T, const float* __restrict__ c3,
    const float* __restrict__ control, const float* __restrict__ war,
    const float* __restrict__ know, float* __restrict__ readw){
  const int b = blockIdx.x, tid=threadIdx.x, lane=tid&31, wave=tid>>5;
  const int lidx=lane&15, half8=(lane>>4)*8;
  __shared__ __align__(16) bf16 hlds[16*512];   // one 16-row n-tile of h, bf16
  __shared__ float cw[512], c3l[512];
  __shared__ float logits[224];
  __shared__ float ar[224];
  for (int i=tid;i<512;i+=256){
    cw[i]  = control[(size_t)b*512+i]*war[i];
    c3l[i] = c3[(size_t)b*512+i];
  }
  if (tid<224) logits[tid]=0.f;
  __syncthreads();

  for (int ntile=0; ntile<13; ++ntile){
    // ---- stage 1: h n-tile (16 x 512), 4 d-tiles per wave, K=512 over e (twice) ----
    v8f acc[4];
    #pragma unroll
    for (int j=0;j<4;++j) acc[j]=vzero8();
    const size_t arow = ((size_t)b*208 + ntile*16 + lidx)*512;
    for (int kt=0; kt<16; ++kt){
      int ko = kt*32 + half8;
      v16bf a1 = ld_op_bf16(pk1T + arow + ko);
      v16bf a2 = ld_op_bf16(pkT  + arow + ko);
      #pragma unroll
      for (int j=0;j<4;++j){
        int dcol = (wave*4+j)*16 + lidx;
        v16bf b1 = ld_op_bf16(Wc1aT + (size_t)dcol*512 + ko);
        acc[j] = WMMA_BF16(a1,b1,acc[j]);
        v16bf b2 = ld_op_bf16(Wc1bT + (size_t)dcol*512 + ko);
        acc[j] = WMMA_BF16(a2,b2,acc[j]);
      }
    }
    #pragma unroll
    for (int j=0;j<4;++j){
      int d = (wave*4+j)*16 + lidx;
      float c3v = c3l[d];
      #pragma unroll
      for (int r=0;r<8;++r){
        float v = acc[j][r] + c3v;
        hlds[(half8 + r)*512 + d] = (bf16)(v > 0.f ? v : 0.f);
      }
    }
    __syncthreads();
    // ---- stage 2: h2 = h@Wc2 (A from LDS) fused into logits (bc2/bar cancel in softmax) ----
    v8f acc2[4];
    #pragma unroll
    for (int j=0;j<4;++j) acc2[j]=vzero8();
    for (int kt=0;kt<16;++kt){
      int ko = kt*32 + half8;
      v16bf a = ld_op_bf16(hlds + lidx*512 + ko);
      #pragma unroll
      for (int j=0;j<4;++j){
        int dp = (wave*4+j)*16 + lidx;
        v16bf bb = ld_op_bf16(Wc2T + (size_t)dp*512 + ko);
        acc2[j] = WMMA_BF16(a,bb,acc2[j]);
      }
    }
    float part[8];
    #pragma unroll
    for (int r=0;r<8;++r) part[r]=0.f;
    #pragma unroll
    for (int j=0;j<4;++j){
      float cwv = cw[(wave*4+j)*16 + lidx];
      #pragma unroll
      for (int r=0;r<8;++r) part[r] += acc2[j][r]*cwv;
    }
    for (int m=1;m<16;m<<=1){
      #pragma unroll
      for (int r=0;r<8;++r) part[r] += __shfl_xor(part[r], m, 32);
    }
    if (lidx==0){
      #pragma unroll
      for (int r=0;r<8;++r) atomicAdd(&logits[ntile*16 + half8 + r], part[r]);
    }
    __syncthreads();
  }
  // mask the 12 padded columns, softmax over 196 (wave 0)
  if (tid>=196 && tid<224) logits[tid] = -1e30f;
  __syncthreads();
  if (wave==0){
    float mx=-1e30f;
    for (int n=lane;n<196;n+=32) mx = fmaxf(mx, logits[n]);
    #pragma unroll
    for (int m=1;m<32;m<<=1) mx = fmaxf(mx, __shfl_xor(mx,m,32));
    float s=0.f;
    for (int n=lane;n<196;n+=32){ float e=__expf(logits[n]-mx); ar[n]=e; s+=e; }
    #pragma unroll
    for (int m=1;m<32;m<<=1) s += __shfl_xor(s,m,32);
    float inv = 1.f/s;
    for (int n=lane;n<196;n+=32) ar[n]*=inv;
  }
  __syncthreads();
  // read[b,d] = sum_n know[b,d,n]*a_r[n] : wave-cooperative per d-row, fully coalesced.
  for (int d = wave; d < 512; d += 8){
    const float* kr = know + ((size_t)b*512 + d)*196;
    float p = 0.f;
    #pragma unroll
    for (int n0 = 0; n0 < 224; n0 += 32){
      int n = n0 + lane;
      if (n < 196) p += kr[n]*ar[n];
    }
    #pragma unroll
    for (int m=1;m<32;m<<=1) p += __shfl_xor(p, m, 32);
    if (lane==0) readw[(size_t)b*512 + d] = p;
  }
}

extern "C" void kernel_launch(void* const* d_in, const int* in_sizes, int n_in,
                              void* d_out, int out_size, void* d_ws, size_t ws_size,
                              hipStream_t stream){
  const float* words    = (const float*)d_in[0];
  const float* question = (const float*)d_in[1];
  const float* know     = (const float*)d_in[2];
  const float* control0 = (const float*)d_in[3];
  const float* memory0  = (const float*)d_in[4];
  const float* Wsc = (const float*)d_in[5];
  const float* bsc = (const float*)d_in[6];
  const float* Wp  = (const float*)d_in[7];
  const float* bp  = (const float*)d_in[8];
  const float* Wcq = (const float*)d_in[9];
  const float* bcq = (const float*)d_in[10];
  const float* wac = (const float*)d_in[11];
  // d_in[12]=bac: constant along softmax axis -> cancels
  const float* Wm  = (const float*)d_in[13];
  const float* bm  = (const float*)d_in[14];
  const float* Wk  = (const float*)d_in[15];
  const float* bk  = (const float*)d_in[16];
  const float* Wc1 = (const float*)d_in[17];
  const float* bc1 = (const float*)d_in[18];
  const float* Wc2 = (const float*)d_in[19];
  // d_in[20]=bc2: constant along read-softmax axis -> cancels
  const float* war = (const float*)d_in[21];
  // d_in[22]=bar cancels; d_in[23..26]=Wwc/bwc/waw/baw: write attention is over a singleton
  // time axis -> softmax==1 -> attn_mem==memory0, whole branch dead.
  const float* Wwcat = (const float*)d_in[27];
  const float* bwcat = (const float*)d_in[28];

  float* outC  = (float*)d_out;          // control (256 x 512)
  float* outNM = outC + 256*512;         // next_mem (256 x 512)

  // workspace layout
  float* wsf   = (float*)d_ws;
  float* qv    = wsf;
  float* pav   = qv    + 131072;
  float* cqv   = pav   + 131072;
  float* pmv   = cqv   + 131072;
  float* c3v   = pmv   + 131072;
  float* readv = c3v   + 131072;
  bf16* wb     = (bf16*)(readv + 131072);
  bf16* WscT   = wb;                    // 512 x 1024
  bf16* WcqTlo = WscT   + 524288;
  bf16* WcqThi = WcqTlo + 262144;
  bf16* WpT    = WcqThi + 262144;
  bf16* WmT    = WpT    + 262144;
  bf16* WkT    = WmT    + 262144;
  bf16* Wc1aT  = WkT    + 262144;
  bf16* Wc1bT  = Wc1aT  + 262144;
  bf16* Wc1cT  = Wc1bT  + 262144;
  bf16* Wc2T   = Wc1cT  + 262144;
  bf16* Wcat0T = Wc2T   + 262144;
  bf16* W12T   = Wcat0T + 262144;
  bf16* pkT    = W12T   + 262144;       // 256 x 208 x 512
  bf16* pk1T   = pkT    + 27262976;

  dim3 blk(256);
  auto Tr = [&](const float* s, const float* s2, bf16* d, int K, int N){
    transpose_cvt_kernel<<<dim3((K*N+255)/256), blk, 0, stream>>>(s, s2, d, K, N);
  };
  Tr(Wsc,            nullptr,          WscT,   1024, 512);
  Tr(Wcq,            nullptr,          WcqTlo,  512, 512);
  Tr(Wcq+512*512,    nullptr,          WcqThi,  512, 512);
  Tr(Wp,             nullptr,          WpT,     512, 512);
  Tr(Wm,             nullptr,          WmT,     512, 512);
  Tr(Wk,             nullptr,          WkT,     512, 512);
  Tr(Wc1,            nullptr,          Wc1aT,   512, 512);
  Tr(Wc1+512*512,    nullptr,          Wc1bT,   512, 512);
  Tr(Wc1+1024*512,   nullptr,          Wc1cT,   512, 512);
  Tr(Wc2,            nullptr,          Wc2T,    512, 512);
  Tr(Wwcat,          nullptr,          Wcat0T,  512, 512);
  Tr(Wwcat+512*512,  Wwcat+1024*512,   W12T,    512, 512);

  // control path
  gemm2_wmma_kernel<<<64,blk,0,stream>>>(question, WscT, 1024, nullptr, nullptr, 0, bsc, qv,  256, 512, 1);
  gemm2_wmma_kernel<<<64,blk,0,stream>>>(qv,       WpT,   512, nullptr, nullptr, 0, bp,  pav, 256, 512, 0);
  gemm2_wmma_kernel<<<64,blk,0,stream>>>(control0, WcqTlo,512, pav, WcqThi, 512,    bcq, cqv, 256, 512, 0);
  ctrl_attn_kernel<<<256,blk,0,stream>>>(words, cqv, wac, outC);

  // memory path
  gemm2_wmma_kernel<<<64,blk,0,stream>>>(memory0, WmT,   512, nullptr, nullptr, 0, bm,  pmv, 256, 512, 0);
  gemm2_wmma_kernel<<<64,blk,0,stream>>>(pmv,     Wc1cT, 512, nullptr, nullptr, 0, bc1, c3v, 256, 512, 0);
  pk_wmma_kernel<<<1024,blk,0,stream>>>(know, WkT, pmv, bk, pkT, pk1T);
  mh_wmma_kernel<<<256,blk,0,stream>>>(pkT, pk1T, Wc1aT, Wc1bT, Wc2T, c3v, outC, war, know, readv);

  // next_mem = read@Wwcat[0:D] + memory0@(Wwcat[D:2D]+Wwcat[2D:3D]) + bwcat
  gemm2_wmma_kernel<<<64,blk,0,stream>>>(readv, Wcat0T, 512, memory0, W12T, 512, bwcat, outNM, 256, 512, 0);
}
